// AttnBlock_24206435680777
// MI455X (gfx1250) — compile-verified, matrix operand order checked by reading, not run on hardware
//
#include <hip/hip_runtime.h>
#include <stdint.h>

typedef __attribute__((ext_vector_type(16))) __bf16 v16bf;
typedef __attribute__((ext_vector_type(8)))  float  v8f;

constexpr int kC   = 512;
constexpr int kSEQ = 8192;   // 8 frames * 32 * 32
constexpr int kNHW = 1024;

// float -> bf16 bits, round-to-nearest-even
__device__ __forceinline__ unsigned short f2bf(float f) {
    union { float f; uint32_t u; } v; v.f = f;
    uint32_t r = v.u + 0x7FFFu + ((v.u >> 16) & 1u);
    return (unsigned short)(r >> 16);
}

// Build a 16x32 bf16 A/B fragment from two contiguous 16-byte chunks.
// CDNA5 layout: lanes 0-15 hold K = {0..7, 16..23}, lanes 16-31 hold
// K = {8..15, 24..31}; caller passes base already offset by (hl*8).
union FragU { uint4 u[2]; v16bf v; };
__device__ __forceinline__ v16bf load_frag(const unsigned short* base) {
    FragU f;
    f.u[0] = *(const uint4*)(base);
    f.u[1] = *(const uint4*)(base + 16);
    return f.v;
}

__device__ __forceinline__ v8f wmma_bf16(v16bf a, v16bf b, v8f c) {
    return __builtin_amdgcn_wmma_f32_16x16x32_bf16(false, a, false, b,
                                                   (short)0, c, false, false);
}

// ---------------------------------------------------------------------------
// 1) RMSNorm over channel dim; x is [C][SEQ] fp32, h is [SEQ][C] bf16
// ---------------------------------------------------------------------------
__global__ __launch_bounds__(256) void k_rmsnorm(const float* __restrict__ x,
                                                 const float* __restrict__ gamma,
                                                 unsigned short* __restrict__ h) {
    const int s = blockIdx.x * 256 + threadIdx.x;
    float ss = 0.f;
    for (int c = 0; c < kC; ++c) { float v = x[c * kSEQ + s]; ss += v * v; }
    const float scale = 22.62741699796952f / fmaxf(sqrtf(ss), 1e-12f); // sqrt(512)/nrm
    for (int c = 0; c < kC; ++c) {
        h[s * kC + c] = f2bf(x[c * kSEQ + s] * scale * gamma[c]);
    }
}

// ---------------------------------------------------------------------------
// 2) fp32 -> bf16 weight conversion ([O][C] row-major kept as-is)
// ---------------------------------------------------------------------------
__global__ __launch_bounds__(256) void k_cvt_w(const float* __restrict__ wq,
                                               const float* __restrict__ wk,
                                               const float* __restrict__ wv,
                                               const float* __restrict__ wo,
                                               unsigned short* __restrict__ oq,
                                               unsigned short* __restrict__ ok,
                                               unsigned short* __restrict__ ov,
                                               unsigned short* __restrict__ oo) {
    const int i = blockIdx.x * 256 + threadIdx.x;
    oq[i] = f2bf(wq[i]);
    ok[i] = f2bf(wk[i]);
    ov[i] = f2bf(wv[i]);
    oo[i] = f2bf(wo[i]);
}

// ---------------------------------------------------------------------------
// 3) QKV projections. grid = (SEQ/16, 1, 3); 8 waves/block; each wave owns a
//    16x64 D-tile (4 accumulators) so the A fragment is reused by 4 WMMAs.
//    V stored transposed [C][SEQ].
// ---------------------------------------------------------------------------
__global__ __launch_bounds__(256) void k_qkv(const unsigned short* __restrict__ h,
                                             const unsigned short* __restrict__ wqb,
                                             const unsigned short* __restrict__ wkb,
                                             const unsigned short* __restrict__ wvb,
                                             const float* __restrict__ bq,
                                             const float* __restrict__ bk,
                                             const float* __restrict__ bv,
                                             unsigned short* __restrict__ qo,
                                             unsigned short* __restrict__ ko,
                                             unsigned short* __restrict__ vo) {
    const int wave = threadIdx.x >> 5;
    const int lane = threadIdx.x & 31;
    const int ln   = lane & 15;
    const int hl   = lane >> 4;
    const int m0   = blockIdx.x * 16;
    const int n0   = wave * 64;                 // 8 waves cover all 512 cols
    const int sel  = blockIdx.z;

    const unsigned short* w = (sel == 0) ? wqb : (sel == 1) ? wkb : wvb;
    const float*          b = (sel == 0) ? bq  : (sel == 1) ? bk  : bv;

    const unsigned short* arow = h + (size_t)(m0 + ln) * kC + hl * 8;
    const unsigned short* br0  = w + (size_t)(n0 +  0 + ln) * kC + hl * 8;
    const unsigned short* br1  = w + (size_t)(n0 + 16 + ln) * kC + hl * 8;
    const unsigned short* br2  = w + (size_t)(n0 + 32 + ln) * kC + hl * 8;
    const unsigned short* br3  = w + (size_t)(n0 + 48 + ln) * kC + hl * 8;

    v8f a0 = {}, a1 = {}, a2 = {}, a3 = {};
#pragma unroll
    for (int k0 = 0; k0 < kC; k0 += 32) {
        const v16bf a = load_frag(arow + k0);   // reused 4x
        a0 = wmma_bf16(a, load_frag(br0 + k0), a0);
        a1 = wmma_bf16(a, load_frag(br1 + k0), a1);
        a2 = wmma_bf16(a, load_frag(br2 + k0), a2);
        a3 = wmma_bf16(a, load_frag(br3 + k0), a3);
    }
#pragma unroll
    for (int t = 0; t < 4; ++t) {
        const v8f acc = (t == 0) ? a0 : (t == 1) ? a1 : (t == 2) ? a2 : a3;
        const int n = n0 + t * 16 + ln;
        const float bias = b[n];
#pragma unroll
        for (int i = 0; i < 8; ++i) {
            const unsigned short r = f2bf(acc[i] + bias);
            const int m = m0 + i + hl * 8;
            if (sel == 2)      vo[(size_t)n * kSEQ + m] = r;   // V transposed
            else if (sel == 1) ko[(size_t)m * kC + n]   = r;
            else               qo[(size_t)m * kC + n]   = r;
        }
    }
}

// ---------------------------------------------------------------------------
// 4) Flash attention, block-causal (frame) mask. One workgroup per 32-query
//    tile (two 16-row WMMA tiles -> every K/V fragment feeds 2 WMMAs, every
//    P fragment feeds 4). Mask is 1024-aligned so 128-key blocks are
//    all-or-nothing: no per-element masking needed.
// ---------------------------------------------------------------------------
constexpr int QROWS   = 32;
constexpr int QSTRIDE = 520;  // 512 + 8 bf16 pad (1040 B rows, 16B aligned)
constexpr int SSTRIDE = 132;  // fp32 scores pad
constexpr int PSTRIDE = 136;  // bf16 P pad (272 B rows, 16B aligned)

__global__ __launch_bounds__(256) void k_attn(const unsigned short* __restrict__ qbuf,
                                              const unsigned short* __restrict__ kbuf,
                                              const unsigned short* __restrict__ vtb,
                                              unsigned short* __restrict__ obuf) {
    __shared__ unsigned short sQ[QROWS * QSTRIDE];
    __shared__ float          sS[QROWS * SSTRIDE];
    __shared__ unsigned short sP[QROWS * PSTRIDE];
    __shared__ float sM[QROWS], sL[QROWS], sFac[QROWS], sInv[QROWS];

    const int tid  = threadIdx.x;
    const int wave = tid >> 5;
    const int lane = tid & 31;
    const int ln   = lane & 15;
    const int hl   = lane >> 4;
    const int m0   = blockIdx.x * QROWS;
    const int fq   = m0 / kNHW;
    const int nkb  = (fq + 1) * (kNHW / 128);   // allowed 128-key blocks

    // Stage 32x512 bf16 Q tile into LDS (2048 uint4 moves, padded rows)
#pragma unroll
    for (int it = 0; it < 8; ++it) {
        const int idx = tid + it * 256;
        const int row = idx >> 6, c16 = idx & 63;
        ((uint4*)(&sQ[row * QSTRIDE]))[c16] =
            ((const uint4*)(qbuf + (size_t)(m0 + row) * kC))[c16];
    }
    if (tid < QROWS) { sM[tid] = -1e30f; sL[tid] = 0.f; }
    __syncthreads();

    v8f o[2][4] = {};
    const float iscl = 0.04419417382415922f;   // 1/sqrt(512)

    for (int kbi = 0; kbi < nkb; ++kbi) {
        const int k0g = kbi * 128;
        const int kt  = k0g + wave * 16;       // this wave's 16 keys

        if (kbi + 1 < nkb) {                   // speculative prefetch of next K block
            __builtin_prefetch(kbuf + (size_t)(kt + 128 + ln) * kC, 0, 3);
        }

        // ---- S = Q K^T : 2 query row-tiles x this wave's 16 keys ----
        v8f s0 = {}, s1 = {};
        const unsigned short* aq0 = &sQ[(     ln) * QSTRIDE + hl * 8];
        const unsigned short* aq1 = &sQ[(16 + ln) * QSTRIDE + hl * 8];
        const unsigned short* bk  = kbuf + (size_t)(kt + ln) * kC + hl * 8;
#pragma unroll
        for (int kc = 0; kc < kC; kc += 32) {
            const v16bf bfr = load_frag(bk + kc);        // reused 2x
            s0 = wmma_bf16(load_frag(aq0 + kc), bfr, s0);
            s1 = wmma_bf16(load_frag(aq1 + kc), bfr, s1);
        }
        {
            const int col = wave * 16 + ln;
#pragma unroll
            for (int i = 0; i < 8; ++i) {
                sS[(     i + hl * 8) * SSTRIDE + col] = s0[i] * iscl;
                sS[(16 + i + hl * 8) * SSTRIDE + col] = s1[i] * iscl;
            }
        }
        __syncthreads();

        // ---- online softmax row update (one thread per query row) ----
        if (tid < QROWS) {
            const float* row = &sS[tid * SSTRIDE];
            float mb = -1e30f;
            for (int j = 0; j < 128; ++j) mb = fmaxf(mb, row[j]);
            const float mold = sM[tid];
            const float mnew = fmaxf(mold, mb);
            const float fac  = __expf(mold - mnew);
            float lsum = 0.f;
            unsigned short* prow = &sP[tid * PSTRIDE];
            for (int j = 0; j < 128; ++j) {
                const float p = __expf(row[j] - mnew);
                lsum += p;
                prow[j] = f2bf(p);
            }
            sM[tid]   = mnew;
            sL[tid]   = sL[tid] * fac + lsum;
            sFac[tid] = fac;
        }
        __syncthreads();

        // ---- rescale O accumulators ----
        float fr0[8], fr1[8];
#pragma unroll
        for (int i = 0; i < 8; ++i) {
            fr0[i] = sFac[     i + hl * 8];
            fr1[i] = sFac[16 + i + hl * 8];
        }
#pragma unroll
        for (int t = 0; t < 4; ++t)
#pragma unroll
            for (int i = 0; i < 8; ++i) {
                o[0][t][i] *= fr0[i];
                o[1][t][i] *= fr1[i];
            }

        // ---- O += P V ; this wave owns output cols [wave*64, wave*64+64) ----
        const int c0 = wave * 64;
        const unsigned short* ap0 = &sP[(     ln) * PSTRIDE + hl * 8];
        const unsigned short* ap1 = &sP[(16 + ln) * PSTRIDE + hl * 8];
#pragma unroll
        for (int kk = 0; kk < 4; ++kk) {
            const v16bf a0 = load_frag(ap0 + kk * 32);   // reused 4x
            const v16bf a1 = load_frag(ap1 + kk * 32);   // reused 4x
            const size_t kpos = (size_t)(k0g + kk * 32 + hl * 8);
#pragma unroll
            for (int t = 0; t < 4; ++t) {
                const v16bf bfr =
                    load_frag(vtb + (size_t)(c0 + t * 16 + ln) * kSEQ + kpos); // reused 2x
                o[0][t] = wmma_bf16(a0, bfr, o[0][t]);
                o[1][t] = wmma_bf16(a1, bfr, o[1][t]);
            }
        }
        __syncthreads();
    }

    if (tid < QROWS) sInv[tid] = 1.0f / sL[tid];
    __syncthreads();

    float iv0[8], iv1[8];
#pragma unroll
    for (int i = 0; i < 8; ++i) {
        iv0[i] = sInv[     i + hl * 8];
        iv1[i] = sInv[16 + i + hl * 8];
    }
    const int c0 = wave * 64;
#pragma unroll
    for (int t = 0; t < 4; ++t)
#pragma unroll
        for (int i = 0; i < 8; ++i) {
            const size_t mA = (size_t)(m0 +      i + hl * 8);
            const size_t mB = (size_t)(m0 + 16 + i + hl * 8);
            obuf[mA * kC + c0 + t * 16 + ln] = f2bf(o[0][t][i] * iv0[i]);
            obuf[mB * kC + c0 + t * 16 + ln] = f2bf(o[1][t][i] * iv1[i]);
        }
}

// ---------------------------------------------------------------------------
// 5) Output projection + residual: out[c][s] = x[c][s] + bo[c] + Wo . o_attn
//    Same 16x64-per-wave blocking as k_qkv.
// ---------------------------------------------------------------------------
__global__ __launch_bounds__(256) void k_proj(const unsigned short* __restrict__ ob,
                                              const unsigned short* __restrict__ wob,
                                              const float* __restrict__ bo,
                                              const float* __restrict__ x,
                                              float* __restrict__ out) {
    const int wave = threadIdx.x >> 5;
    const int lane = threadIdx.x & 31;
    const int ln   = lane & 15;
    const int hl   = lane >> 4;
    const int m0   = blockIdx.x * 16;
    const int n0   = wave * 64;

    const unsigned short* arow = ob  + (size_t)(m0 + ln) * kC + hl * 8;
    const unsigned short* br0  = wob + (size_t)(n0 +  0 + ln) * kC + hl * 8;
    const unsigned short* br1  = wob + (size_t)(n0 + 16 + ln) * kC + hl * 8;
    const unsigned short* br2  = wob + (size_t)(n0 + 32 + ln) * kC + hl * 8;
    const unsigned short* br3  = wob + (size_t)(n0 + 48 + ln) * kC + hl * 8;

    v8f a0 = {}, a1 = {}, a2 = {}, a3 = {};
#pragma unroll
    for (int k0 = 0; k0 < kC; k0 += 32) {
        const v16bf a = load_frag(arow + k0);   // reused 4x
        a0 = wmma_bf16(a, load_frag(br0 + k0), a0);
        a1 = wmma_bf16(a, load_frag(br1 + k0), a1);
        a2 = wmma_bf16(a, load_frag(br2 + k0), a2);
        a3 = wmma_bf16(a, load_frag(br3 + k0), a3);
    }
#pragma unroll
    for (int t = 0; t < 4; ++t) {
        const v8f acc = (t == 0) ? a0 : (t == 1) ? a1 : (t == 2) ? a2 : a3;
        const int n = n0 + t * 16 + ln;
        const float bias = bo[n];
#pragma unroll
        for (int i = 0; i < 8; ++i) {
            const int m = m0 + i + hl * 8;
            out[(size_t)n * kSEQ + m] = x[(size_t)n * kSEQ + m] + bias + acc[i];
        }
    }
}

// ---------------------------------------------------------------------------
extern "C" void kernel_launch(void* const* d_in, const int* in_sizes, int n_in,
                              void* d_out, int out_size, void* d_ws, size_t ws_size,
                              hipStream_t stream) {
    const float* x     = (const float*)d_in[0];
    const float* gamma = (const float*)d_in[1];
    const float* wq    = (const float*)d_in[2];
    const float* bq    = (const float*)d_in[3];
    const float* wk    = (const float*)d_in[4];
    const float* bk    = (const float*)d_in[5];
    const float* wv    = (const float*)d_in[6];
    const float* bv    = (const float*)d_in[7];
    const float* wo    = (const float*)d_in[8];
    const float* bo    = (const float*)d_in[9];
    float* out = (float*)d_out;

    unsigned short* ws = (unsigned short*)d_ws;
    const size_t MAT = (size_t)kSEQ * kC;       // 4M bf16 elems = 8 MB
    unsigned short* h   = ws;                   //  0 MB
    unsigned short* qb  = ws + 1 * MAT;         //  8 MB
    unsigned short* kb  = ws + 2 * MAT;         // 16 MB
    unsigned short* vtb = ws + 3 * MAT;         // 24 MB (transposed V)
    unsigned short* ob  = ws + 4 * MAT;         // 32 MB
    unsigned short* wqb = ws + 5 * MAT;         // 40 MB, 4 x 512KB bf16 weights
    unsigned short* wkb = wqb + (size_t)kC * kC;
    unsigned short* wvb = wkb + (size_t)kC * kC;
    unsigned short* wob = wvb + (size_t)kC * kC;

    k_rmsnorm<<<kSEQ / 256, 256, 0, stream>>>(x, gamma, h);
    k_cvt_w<<<(kC * kC) / 256, 256, 0, stream>>>(wq, wk, wv, wo, wqb, wkb, wvb, wob);
    k_qkv<<<dim3(kSEQ / 16, 1, 3), 256, 0, stream>>>(
        h, wqb, wkb, wvb, bq, bk, bv, qb, kb, vtb);
    k_attn<<<kSEQ / QROWS, 256, 0, stream>>>(qb, kb, vtb, ob);
    k_proj<<<dim3(kSEQ / 16, 1), 256, 0, stream>>>(ob, wob, bo, x, out);
}
